// GCNLPA_4621384810949
// MI455X (gfx1250) — matrix operand order repaired and checked by reading, compiled, not run on hardware
//
#include <hip/hip_runtime.h>
#include <stdint.h>

// ---------------------------------------------------------------------------
// Types
// ---------------------------------------------------------------------------
typedef __attribute__((ext_vector_type(16))) __bf16        v16bf;
typedef __attribute__((ext_vector_type(2)))  __bf16        v2bf;
typedef __attribute__((ext_vector_type(8)))  float         v8f;
typedef __attribute__((ext_vector_type(4)))  unsigned int  v4u;
typedef __attribute__((ext_vector_type(4)))  int           v4i;

union Frag {
  v16bf bf;
  v4u   q[2];
};

#define NROWS 8192

#if __has_builtin(__builtin_amdgcn_global_load_async_to_lds_b128) && __has_builtin(__builtin_amdgcn_s_wait_asynccnt)
#define HAS_ASYNC 1
#else
#define HAS_ASYNC 0
#endif

// native converts -> V_CVT_PK_BF16_F32 on gfx1250 (RNE, same as manual rounding)
__device__ __forceinline__ unsigned int pack2bf(float a, float b) {
  v2bf t;
  t.x = (__bf16)a;
  t.y = (__bf16)b;
  return __builtin_bit_cast(unsigned int, t);
}

__device__ __forceinline__ unsigned short f2bf(float f) {
  __bf16 h = (__bf16)f;
  return __builtin_bit_cast(unsigned short, h);
}

__device__ __forceinline__ void copy_row32_bf16(const unsigned short* __restrict__ gp,
                                                unsigned short* __restrict__ lp) {
  // 32 bf16 = 64 bytes = 4 x b128
#if HAS_ASYNC
  __attribute__((address_space(1))) v4i* g = (__attribute__((address_space(1))) v4i*)gp;
  __attribute__((address_space(3))) v4i* l = (__attribute__((address_space(3))) v4i*)lp;
  __builtin_amdgcn_global_load_async_to_lds_b128(g, l, 0, 0);
  __builtin_amdgcn_global_load_async_to_lds_b128(g, l, 16, 0);
  __builtin_amdgcn_global_load_async_to_lds_b128(g, l, 32, 0);
  __builtin_amdgcn_global_load_async_to_lds_b128(g, l, 48, 0);
#else
  const v4u* s = (const v4u*)gp;
  v4u*       d = (v4u*)lp;
  d[0] = s[0]; d[1] = s[1]; d[2] = s[2]; d[3] = s[3];
#endif
}

__device__ __forceinline__ void wait_async_then_barrier() {
#if HAS_ASYNC
  __builtin_amdgcn_s_wait_asynccnt(0);
#endif
  __syncthreads();
}

// ---------------------------------------------------------------------------
// Fused layer kernel:  out = (diag(1/rownorm(|adj*mask|)) * (adj*mask)) @ B^T
//                      + bias (cols < Pbias) , optional relu (cols < Pbias)
// B stored transposed & bf16: Bt[p][k], p in [0, PTILES*16), k in [0, 8192).
// PTILES must be a multiple of 4 (zero-padded) so every wave gets exactly
// PTILES/4 tiles -> no guards around WMMA, EXEC all-1s, no acc phi copies.
// Block: 256 threads (8 waves), 32 output rows per block, all PTILES columns.
// ---------------------------------------------------------------------------
template <int PTILES>
__global__ __launch_bounds__(256)
void gcn_layer_kernel(const float* __restrict__ adj, const float* __restrict__ msk,
                      const unsigned short* __restrict__ Bt, float* __restrict__ out,
                      const float* __restrict__ bias, int Pbias, int doRelu) {
  static_assert((PTILES & 3) == 0, "PTILES must be a multiple of 4");
  constexpr int NT = PTILES / 4;     // col tiles per wave (cgrp stride 4)
  constexpr int Pp = PTILES * 16;

  __shared__ unsigned short lsA[32][40];        // 32 rows x 32 k (padded)
  __shared__ unsigned short lsB[PTILES * 16][32];
  __shared__ float rnorm[32];

  const int t    = threadIdx.x;
  const int ib   = blockIdx.x * 32;
  const int lane = t & 31;
  const int w    = t >> 5;
  const int rsub = w & 1;       // which 16-row half
  const int cgrp = w >> 1;      // column-tile group (0..3)

  if (t < 32) rnorm[t] = 0.0f;

  // cooperative A-load coords (fixed row per thread across whole K loop)
  const int ar = t >> 3;              // 0..31
  const int ac = (t & 7) * 4;         // 0,4,...,28
  const float* adjp = adj + (size_t)(ib + ar) * NROWS + ac;
  const float* mskp = msk + (size_t)(ib + ar) * NROWS + ac;
  float s = 0.0f;

  // WMMA fragment coords (wave32 layouts per CDNA5 ISA 7.12.2)
  const int fm = lane & 15;           // A row within tile
  const int kh = (lane >> 4) * 8;     // A: K chunks {kh..kh+7, kh+16..kh+23}
  const int fn = lane & 15;           // B column within tile
  const int kb = (lane >> 4) * 8;     // B: K chunks {kb..kb+7, kb+16..kb+23}

  v8f acc[NT] = {};

  // software pipeline: global loads for chunk kc are issued one iter ahead
  float4 a4 = *(const float4*)(adjp);
  float4 m4 = *(const float4*)(mskp);

  __syncthreads();

  for (int kc = 0; kc < NROWS; kc += 32) {
    // ---- stage current A tile (fp32 adj*mask -> bf16) + rownorm partial ----
    float p0 = a4.x * m4.x, p1 = a4.y * m4.y, p2 = a4.z * m4.z, p3 = a4.w * m4.w;
    s += fabsf(p0) + fabsf(p1) + fabsf(p2) + fabsf(p3);
    *(unsigned int*)&lsA[ar][ac]     = pack2bf(p0, p1);
    *(unsigned int*)&lsA[ar][ac + 2] = pack2bf(p2, p3);

    // ---- stage B chunk (bf16, already transposed in global) ----
    for (int p = t; p < Pp; p += 256)
      copy_row32_bf16(Bt + (size_t)p * NROWS + kc, &lsB[p][0]);

    wait_async_then_barrier();

    // ---- prefetch next A chunk from HBM (overlaps with WMMA below) ----
    if (kc + 32 < NROWS) {
      a4 = *(const float4*)(adjp + kc + 32);
      m4 = *(const float4*)(mskp + kc + 32);
    }

    // ---- WMMA ----
    Frag af;
    af.q[0] = *(const v4u*)&lsA[rsub * 16 + fm][kh];
    af.q[1] = *(const v4u*)&lsA[rsub * 16 + fm][kh + 16];
#pragma unroll
    for (int nt = 0; nt < NT; ++nt) {
      const int ct = cgrp + nt * 4;
      Frag bf_;
      bf_.q[0] = *(const v4u*)&lsB[ct * 16 + fn][kb];
      bf_.q[1] = *(const v4u*)&lsB[ct * 16 + fn][kb + 16];
      acc[nt] = __builtin_amdgcn_wmma_f32_16x16x32_bf16(
          false, af.bf, false, bf_.bf, (short)0, acc[nt], false, false);
    }
    __syncthreads();
  }

  atomicAdd(&rnorm[ar], s);
  __syncthreads();

  // ---- epilogue: /rownorm, +bias, relu, store ----
  const int mrow0 = (lane >> 4) * 8;
#pragma unroll
  for (int nt = 0; nt < NT; ++nt) {
    const int ct   = cgrp + nt * 4;
    const int col  = ct * 16 + fn;
    const float ba = (col < Pbias) ? bias[col] : 0.0f;
    const bool rl  = doRelu && (col < Pbias);
#pragma unroll
    for (int i = 0; i < 8; ++i) {
      const int m   = rsub * 16 + mrow0 + i;
      const float n = fmaxf(rnorm[m], 1e-12f);
      float v = acc[nt][i] / n + ba;
      if (rl) v = fmaxf(v, 0.0f);
      out[(size_t)(ib + m) * Pp + col] = v;
    }
  }
}

// ---------------------------------------------------------------------------
// Generic WMMA GEMM: C[8192 x PTILES*16] = A(fp32, lda) @ Wt^T (bf16 Pp x K)
// PTILES multiple of 4.
// ---------------------------------------------------------------------------
template <int PTILES>
__global__ __launch_bounds__(256)
void gemm_bf16_kernel(const float* __restrict__ A, int lda, int K,
                      const unsigned short* __restrict__ Wt, float* __restrict__ C) {
  static_assert((PTILES & 3) == 0, "PTILES must be a multiple of 4");
  constexpr int NT = PTILES / 4;
  constexpr int Pp = PTILES * 16;

  __shared__ unsigned short lsA[32][40];
  __shared__ unsigned short lsB[PTILES * 16][32];

  const int t    = threadIdx.x;
  const int ib   = blockIdx.x * 32;
  const int lane = t & 31;
  const int w    = t >> 5;
  const int rsub = w & 1;
  const int cgrp = w >> 1;

  const int ar = t >> 3;
  const int ac = (t & 7) * 4;
  const float* Ap = A + (size_t)(ib + ar) * lda + ac;

  const int fm = lane & 15;
  const int kh = (lane >> 4) * 8;
  const int fn = lane & 15;
  const int kb = (lane >> 4) * 8;

  v8f acc[NT] = {};

  float4 a4 = *(const float4*)(Ap);

  for (int kc = 0; kc < K; kc += 32) {
    *(unsigned int*)&lsA[ar][ac]     = pack2bf(a4.x, a4.y);
    *(unsigned int*)&lsA[ar][ac + 2] = pack2bf(a4.z, a4.w);

    for (int p = t; p < Pp; p += 256)
      copy_row32_bf16(Wt + (size_t)p * K + kc, &lsB[p][0]);

    wait_async_then_barrier();

    if (kc + 32 < K) a4 = *(const float4*)(Ap + kc + 32);

    Frag af;
    af.q[0] = *(const v4u*)&lsA[rsub * 16 + fm][kh];
    af.q[1] = *(const v4u*)&lsA[rsub * 16 + fm][kh + 16];
#pragma unroll
    for (int nt = 0; nt < NT; ++nt) {
      const int ct = cgrp + nt * 4;
      Frag bf_;
      bf_.q[0] = *(const v4u*)&lsB[ct * 16 + fn][kb];
      bf_.q[1] = *(const v4u*)&lsB[ct * 16 + fn][kb + 16];
      acc[nt] = __builtin_amdgcn_wmma_f32_16x16x32_bf16(
          false, af.bf, false, bf_.bf, (short)0, acc[nt], false, false);
    }
    __syncthreads();
  }

  const int mrow0 = (lane >> 4) * 8;
#pragma unroll
  for (int nt = 0; nt < NT; ++nt) {
    const int ct  = cgrp + nt * 4;
    const int col = ct * 16 + fn;
#pragma unroll
    for (int i = 0; i < 8; ++i) {
      const int m = rsub * 16 + mrow0 + i;
      C[(size_t)(ib + m) * Pp + col] = acc[nt][i];
    }
  }
}

// ---------------------------------------------------------------------------
// Pack kernels (transpose + fp32->bf16, zero-pad to tile multiples)
// ---------------------------------------------------------------------------
__global__ void pack_wt_kernel(const float* __restrict__ W, unsigned short* __restrict__ Wt,
                               int K, int P, int Pp) {
  int idx = blockIdx.x * 256 + threadIdx.x;
  if (idx >= Pp * K) return;
  int p = idx / K, k = idx - p * K;
  Wt[idx] = (p < P) ? f2bf(W[(size_t)k * P + p]) : (unsigned short)0;
}

// Bt1[p][i], Pp=320: p<256 -> support1[i][p] ; 256<=p<296 -> y[i][p-256] ; else 0
__global__ void pack_bt1_kernel(const float* __restrict__ sup, const float* __restrict__ y,
                                unsigned short* __restrict__ Bt) {
  int idx = blockIdx.x * 256 + threadIdx.x;   // 320*8192 threads
  int p = idx >> 13, i = idx & 8191;
  unsigned short v;
  if (p < 256)      v = f2bf(sup[(size_t)i * 256 + p]);
  else if (p < 296) v = f2bf(y[(size_t)i * 40 + (p - 256)]);
  else              v = 0;
  Bt[idx] = v;
}

// Bt2[p][i], Pp=128: p<40 -> support2[i][p] (ld 64) ;
//                    40<=p<80 -> yhat1 = out1[i][256+(p-40)] (ld 320) ; else 0
__global__ void pack_bt2_kernel(const float* __restrict__ sup2, const float* __restrict__ out1,
                                unsigned short* __restrict__ Bt) {
  int idx = blockIdx.x * 256 + threadIdx.x;   // 128*8192 threads
  int p = idx >> 13, i = idx & 8191;
  unsigned short v;
  if (p < 40)      v = f2bf(sup2[(size_t)i * 64 + p]);
  else if (p < 80) v = f2bf(out1[(size_t)i * 320 + 256 + (p - 40)]);
  else             v = 0;
  Bt[idx] = v;
}

// ---------------------------------------------------------------------------
// Row-wise log-softmax over 40 cols: wave 0 -> h2 (cols 0..39),
// wave 1 -> yhat2 (cols 40..79) of out2 (ld 128). Block = 64 threads.
// ---------------------------------------------------------------------------
__global__ __launch_bounds__(64)
void logsoftmax_kernel(const float* __restrict__ out2, float* __restrict__ dst) {
  const int row  = blockIdx.x;
  const int w    = threadIdx.x >> 5;
  const int lane = threadIdx.x & 31;
  const float* src = out2 + (size_t)row * 128 + w * 40;

  float v0 = src[lane];                                            // lanes 0..31 < 40
  float v1 = (lane + 32 < 40) ? src[lane + 32] : -__builtin_inff();
  float m = fmaxf(v0, v1);
#pragma unroll
  for (int o = 16; o > 0; o >>= 1) m = fmaxf(m, __shfl_xor(m, o, 32));
  float e = __expf(v0 - m);
  if (lane + 32 < 40) e += __expf(v1 - m);
#pragma unroll
  for (int o = 16; o > 0; o >>= 1) e += __shfl_xor(e, o, 32);
  const float l = __logf(e);

  float* d = dst + (size_t)w * (8192 * 40) + (size_t)row * 40;
  d[lane] = v0 - m - l;
  if (lane + 32 < 40) d[lane + 32] = v1 - m - l;
}

// ---------------------------------------------------------------------------
// Host launcher
// ---------------------------------------------------------------------------
extern "C" void kernel_launch(void* const* d_in, const int* in_sizes, int n_in,
                              void* d_out, int out_size, void* d_ws, size_t ws_size,
                              hipStream_t stream) {
  const float* x     = (const float*)d_in[0];
  const float* adj   = (const float*)d_in[1];
  const float* y     = (const float*)d_in[2];
  const float* mask1 = (const float*)d_in[3];
  const float* mask2 = (const float*)d_in[4];
  const float* w1    = (const float*)d_in[5];
  const float* b1    = (const float*)d_in[6];
  const float* w2    = (const float*)d_in[7];
  const float* b2    = (const float*)d_in[8];
  (void)in_sizes; (void)n_in; (void)out_size; (void)ws_size;

  size_t off = 0;
  auto take = [&](size_t bytes) {
    void* p = (char*)d_ws + off;
    off += (bytes + 255) & ~(size_t)255;
    return p;
  };
  float*          sup1 = (float*)         take(8192ull * 256 * 4);
  unsigned short* Bt1  = (unsigned short*)take(320ull * 8192 * 2);
  float*          out1 = (float*)         take(8192ull * 320 * 4);
  float*          sup2 = (float*)         take(8192ull * 64 * 4);
  unsigned short* Bt2  = (unsigned short*)take(128ull * 8192 * 2);
  float*          out2 = (float*)         take(8192ull * 128 * 4);
  unsigned short* Wt1  = (unsigned short*)take(256ull * 512 * 2);
  unsigned short* Wt2  = (unsigned short*)take(64ull * 256 * 2);

  // weight transpose/convert (tiny, L2 resident)
  pack_wt_kernel<<<(256 * 512) / 256, 256, 0, stream>>>(w1, Wt1, 512, 256, 256);
  pack_wt_kernel<<<(64 * 256) / 256, 256, 0, stream>>>(w2, Wt2, 256, 40, 64);

  // support1 = x @ w1   (8192x512 @ 512x256)
  gemm_bf16_kernel<16><<<256, 256, 0, stream>>>(x, 512, 512, Wt1, sup1);

  // B1 = [support1 | y | pad] transposed bf16 (320 x 8192)
  pack_bt1_kernel<<<(320 * 8192) / 256, 256, 0, stream>>>(sup1, y, Bt1);

  // layer 1: out1 = norm(adj*mask1) @ B1  (+b1, relu on first 256 cols)
  gcn_layer_kernel<20><<<256, 256, 0, stream>>>(adj, mask1, Bt1, out1, b1, 256, 1);

  // support2 = h1 @ w2   (h1 = out1 cols 0..255 ld 320, already relu'd)
  gemm_bf16_kernel<4><<<256, 256, 0, stream>>>(out1, 320, 256, Wt2, sup2);

  // B2 = [support2 | yhat1 | pad] transposed bf16 (128 x 8192)
  pack_bt2_kernel<<<(128 * 8192) / 256, 256, 0, stream>>>(sup2, out1, Bt2);

  // layer 2: out2 = norm(adj*mask2) @ B2  (+b2 on first 40 cols, no relu)
  gcn_layer_kernel<8><<<256, 256, 0, stream>>>(adj, mask2, Bt2, out2, b2, 40, 0);

  // log_softmax over h2 and yhat2 -> d_out
  logsoftmax_kernel<<<8192, 64, 0, stream>>>(out2, (float*)d_out);
}
